// NewVectorQuantizer_38431367365340
// MI455X (gfx1250) — matrix operand order repaired and checked by reading, not compiled, not run on hardware
//
#include <hip/hip_runtime.h>

typedef __attribute__((ext_vector_type(16))) __bf16        v16bf;
typedef __attribute__((ext_vector_type(8)))  float         v8f;
typedef __attribute__((ext_vector_type(8)))  unsigned int  v8u;

#define BDIM   32
#define CDIM   64
#define HDIM   64
#define WDIM   64
#define NEMB   512
#define PLANE  (HDIM*WDIM)            // 4096
#define IMGSZ  (CDIM*HDIM*WDIM)       // 262144
#define OUTPLANE (BDIM*IMGSZ)         // 8388608 floats per output tensor
#define NTILES ((BDIM*HDIM*WDIM)/16)  // 8192 token tiles

__device__ __forceinline__ unsigned f2bf(float f) {
  union { float f; unsigned u; } v; v.f = f;
  unsigned u = v.u;
  u += 0x7FFFu + ((u >> 16) & 1u);    // round-to-nearest-even
  return (u >> 16) & 0xFFFFu;
}

__global__ __launch_bounds__(256) void vq_argmin_wmma(
    const float* __restrict__ x,      // [B,C,H,W]
    const float* __restrict__ embed,  // [C, NEMB]
    float* __restrict__ out)          // [qwg | q | ind] flat
{
  // Codebook staged as bf16 K-pairs already in WMMA B-operand order:
  // pkB[k2 * 512 + n] = {bf16(embed[2*k2][n]), bf16(embed[2*k2+1][n])}
  __shared__ unsigned pkB[32 * NEMB];   // 64 KB
  __shared__ float    enorm[NEMB];      // 2 KB
  __shared__ int      tokIdxS[8][16];   // 0.5 KB

  const int tid = threadIdx.x;

  for (int i = tid; i < 32 * NEMB; i += 256) {
    int k2 = i >> 9, n = i & 511;
    float e0 = embed[(2 * k2)     * NEMB + n];
    float e1 = embed[(2 * k2 + 1) * NEMB + n];
    pkB[i] = f2bf(e0) | (f2bf(e1) << 16);
  }
  for (int e = tid; e < NEMB; e += 256) {
    float s = 0.f;
    #pragma unroll 8
    for (int c = 0; c < CDIM; ++c) { float v = embed[c * NEMB + e]; s += v * v; }
    enorm[e] = s;
  }
  __syncthreads();

  const int wave = tid >> 5;
  const int lane = tid & 31;
  const int m    = lane & 15;   // row (token) within tile / column within B
  const int g    = lane >> 4;   // half-wave group

  const int tile = blockIdx.x * 8 + wave;     // 0..8191
  const int tok0 = tile << 4;                 // first token index (== b*4096+h*64+w0)
  const int b    = tok0 >> 12;
  const int h    = (tok0 >> 6) & 63;
  const int w0   = tok0 & 63;                 // multiple of 16

  const float* xb = x + (size_t)b * IMGSZ + h * WDIM + w0;

  // ---- Build A operands (16 tokens x 64 features) per 16-bit A layout:
  // lane m = row M; VGPR v holds K = {kv, kv+1}, kv = 8g+2v (v<4) / 8g+2v+8 (v>=4)
  v8u a0u, a1u;
  #pragma unroll
  for (int v = 0; v < 8; ++v) {
    int k = (v < 4) ? (8 * g + 2 * v) : (8 * g + 2 * v + 8);
    float f0 = xb[(k)      * PLANE + m];
    float f1 = xb[(k + 1)  * PLANE + m];
    a0u[v] = f2bf(f0) | (f2bf(f1) << 16);
    float f2 = xb[(k + 32) * PLANE + m];
    float f3 = xb[(k + 33) * PLANE + m];
    a1u[v] = f2bf(f2) | (f2bf(f3) << 16);
  }
  const v16bf A0 = __builtin_bit_cast(v16bf, a0u);   // K = 0..31
  const v16bf A1 = __builtin_bit_cast(v16bf, a1u);   // K = 32..63

  // ---- Fused GEMM + argmax over 32 embed tiles of 16
  float best[8];
  int   bidx[8];
  #pragma unroll
  for (int j = 0; j < 8; ++j) { best[j] = -3.4e38f; bidx[j] = 0; }

  for (int t = 0; t < 32; ++t) {
    const int n = (t << 4) + m;          // embedding column this lane owns
    v8u blu, bhu;
    #pragma unroll
    for (int v = 0; v < 8; ++v) {        // B layout: VGPR v ↔ K pair 8g+2v (v<4) / 8g+2v+8 (v>=4)
      int k2 = (v < 4) ? (v + 4 * g) : (v + 4 * g + 4);
      blu[v] = pkB[(k2 << 9) + n];
      bhu[v] = pkB[((k2 + 16) << 9) + n];
    }
    v8f c = {};
    c = __builtin_amdgcn_wmma_f32_16x16x32_bf16(
            false, A0, false, __builtin_bit_cast(v16bf, blu), (short)0, c, false, false);
    c = __builtin_amdgcn_wmma_f32_16x16x32_bf16(
            false, A1, false, __builtin_bit_cast(v16bf, bhu), (short)0, c, false, false);

    const float en = enorm[n];
    #pragma unroll
    for (int j = 0; j < 8; ++j) {        // D layout: VGPR j ↔ row M = j + 8g
      float val = 2.f * c[j] - en;       // argmax(2 f·e − ||e||²) == argmin dist
      if (val > best[j]) { best[j] = val; bidx[j] = n; }
    }
  }

  // ---- Reduce across the 16 lanes sharing each row (xor within half-wave)
  #pragma unroll
  for (int off = 1; off < 16; off <<= 1) {
    #pragma unroll
    for (int j = 0; j < 8; ++j) {
      float ob = __shfl_xor(best[j], off, 32);
      int   oi = __shfl_xor(bidx[j], off, 32);
      if (ob > best[j] || (ob == best[j] && oi < bidx[j])) { best[j] = ob; bidx[j] = oi; }
    }
  }
  if (m == 0) {
    #pragma unroll
    for (int j = 0; j < 8; ++j) tokIdxS[wave][g * 8 + j] = bidx[j];
  }
  __builtin_amdgcn_wave_barrier();

  // ---- Gather codebook column + scatter to both NCHW outputs
  const int   ti   = tokIdxS[wave][m];
  if (g == 0) out[2 * OUTPLANE + tok0 + m] = (float)ti;   // embed_ind (as float)

  const float* ecol = embed + ti;                          // embed[c*512 + ti]
  float* ob = out + (g ? OUTPLANE : 0)
                  + (size_t)b * IMGSZ + h * WDIM + w0 + m;
  #pragma unroll 8
  for (int c = 0; c < CDIM; ++c) {
    ob[c * PLANE] = ecol[c << 9];
  }
}

extern "C" void kernel_launch(void* const* d_in, const int* in_sizes, int n_in,
                              void* d_out, int out_size, void* d_ws, size_t ws_size,
                              hipStream_t stream) {
  const float* x     = (const float*)d_in[0];  // [32,64,64,64]
  const float* embed = (const float*)d_in[1];  // [64,512]
  float* out         = (float*)d_out;          // 2*8388608 + 131072 floats
  (void)in_sizes; (void)n_in; (void)out_size; (void)d_ws; (void)ws_size;

  vq_argmin_wmma<<<NTILES / 8, 256, 0, stream>>>(x, embed, out);
}
